// MA_73478300500338
// MI455X (gfx1250) — matrix operand order repaired and checked by reading, not compiled, not run on hardware
//
#include <hip/hip_runtime.h>
#include <hip/hip_bf16.h>
#include <math.h>

// ---------------------------------------------------------------------------
// Problem constants (match reference)
// ---------------------------------------------------------------------------
#define BQ   256      // batch
#define NK   100000   // memory keys
#define DD   512      // embedding dim
#define AU_  256      // attention units
#define CC   100      // classes
#define KK   32       // top-k

typedef __attribute__((ext_vector_type(16))) _Float16 v16h;
typedef __attribute__((ext_vector_type(8)))  _Float16 v8h;
typedef __attribute__((ext_vector_type(8)))  float    v8f;

// ---------------------------------------------------------------------------
// WMMA fragment loaders (wave32, 16x16x32 f16, f32 accum)
// A layout: lane m=lane&15 holds row m; hi=lane>>4 selects K run {0..15}
//           (hi=0) or {16..31} (hi=1).
// B layout: lane n=lane&15 holds col n; hi selects K run {0..15} / {16..31}.
// C layout: element v of v8f: M = v + (lane>>4)*8, N = lane&15.
// ---------------------------------------------------------------------------
__device__ __forceinline__ v16h load_a_frag(const _Float16* As, int rowBase,
                                            int strideH, int lane) {
    const int m = lane & 15, hi = lane >> 4;
    const _Float16* r = As + (rowBase + m) * strideH;
    v8h lo = *(const v8h*)(r + hi * 16);
    v8h hv = *(const v8h*)(r + hi * 16 + 8);
    return __builtin_shufflevector(lo, hv, 0,1,2,3,4,5,6,7,8,9,10,11,12,13,14,15);
}

__device__ __forceinline__ v16h load_b_frag(const _Float16* Bs, int colBase,
                                            int strideH, int lane) {
    const int n = lane & 15, koff = (lane >> 4) * 16;
    return *(const v16h*)(Bs + (colBase + n) * strideH + koff);
}

// ---------------------------------------------------------------------------
// 1) q = relu(query); qh = f16(q); qn = ||q||
// ---------------------------------------------------------------------------
__global__ void prep_q_kernel(const float* __restrict__ query,
                              float* __restrict__ qrelu,
                              _Float16* __restrict__ qh,
                              float* __restrict__ qn) {
    __shared__ float red[128];
    const int b = blockIdx.x, t = threadIdx.x;
    float ss = 0.f;
    for (int i = t; i < DD; i += 128) {
        float v = fmaxf(query[b * DD + i], 0.f);
        qrelu[b * DD + i] = v;
        qh[b * DD + i] = (_Float16)v;
        ss += v * v;
    }
    red[t] = ss; __syncthreads();
    for (int s = 64; s > 0; s >>= 1) { if (t < s) red[t] += red[t + s]; __syncthreads(); }
    if (t == 0) qn[b] = sqrtf(red[0]);
}

// ---------------------------------------------------------------------------
// 2) mn[n] = ||memory_keys[n]||  (one wave per row, coalesced b128 loads)
// ---------------------------------------------------------------------------
__global__ void key_norms_kernel(const float* __restrict__ mk,
                                 float* __restrict__ mn) {
    const int lane = threadIdx.x & 31;
    const int wid  = (blockIdx.x * blockDim.x + threadIdx.x) >> 5;
    const int nw   = (gridDim.x * blockDim.x) >> 5;
    for (int row = wid; row < NK; row += nw) {
        const float4* p = (const float4*)(mk + (long long)row * DD + lane * 16);
        float ss = 0.f;
        #pragma unroll
        for (int j = 0; j < 4; ++j) {
            float4 v = p[j];
            ss += v.x * v.x + v.y * v.y + v.z * v.z + v.w * v.w;
        }
        #pragma unroll
        for (int off = 16; off > 0; off >>= 1) ss += __shfl_xor(ss, off, 32);
        if (lane == 0) mn[row] = sqrtf(ss);
    }
}

// ---------------------------------------------------------------------------
// 3) sim = (q @ mk^T) / max(qn*mn, eps)  via v_wmma_f32_16x16x32_f16
//    Workgroup: 256 rows x 32 key-cols; memory_keys read exactly once.
// ---------------------------------------------------------------------------
__global__ void sim_gemm_kernel(const _Float16* __restrict__ qh,
                                const float* __restrict__ mk,
                                const float* __restrict__ qn,
                                const float* __restrict__ mn,
                                float* __restrict__ sim) {
    __shared__ __attribute__((aligned(64))) _Float16 As[256 * 32]; // [row][k]
    __shared__ __attribute__((aligned(64))) _Float16 Bs[32 * 32];  // [keycol][k]
    const int t = threadIdx.x, lane = t & 31, wave = t >> 5;
    const int n0 = blockIdx.x * 32;

    v8f acc[2][2] = {};

    for (int k0 = 0; k0 < DD; k0 += 32) {
        __syncthreads();
        // A tile: thread t copies 32 halves of row t
        {
            const v8h* s = (const v8h*)(qh + t * DD + k0);
            v8h* d = (v8h*)(As + t * 32);
            d[0] = s[0]; d[1] = s[1]; d[2] = s[2]; d[3] = s[3];
        }
        // B tile: f32 -> f16 convert on the fly (keys are row-major over k)
        {
            const int n = t >> 3, kk = (t & 7) * 4;
            float4 v = *(const float4*)(mk + (long long)(n0 + n) * DD + k0 + kk);
            _Float16* d = Bs + n * 32 + kk;
            d[0] = (_Float16)v.x; d[1] = (_Float16)v.y;
            d[2] = (_Float16)v.z; d[3] = (_Float16)v.w;
        }
        __syncthreads();
        #pragma unroll
        for (int rt = 0; rt < 2; ++rt) {
            v16h a = load_a_frag(As, wave * 32 + rt * 16, 32, lane);
            #pragma unroll
            for (int ct = 0; ct < 2; ++ct) {
                v16h b = load_b_frag(Bs, ct * 16, 32, lane);
                acc[rt][ct] = __builtin_amdgcn_wmma_f32_16x16x32_f16(
                    false, a, false, b, (short)0, acc[rt][ct], false, false);
            }
        }
    }

    const int nn = lane & 15, hi = lane >> 4;
    #pragma unroll
    for (int rt = 0; rt < 2; ++rt)
        #pragma unroll
        for (int ct = 0; ct < 2; ++ct)
            #pragma unroll
            for (int v = 0; v < 8; ++v) {
                const int row = wave * 32 + rt * 16 + hi * 8 + v;
                const int col = n0 + ct * 16 + nn;
                const float den = fmaxf(qn[row] * mn[col], 1e-8f);
                sim[(long long)row * NK + col] = acc[rt][ct][v] / den;
            }
}

// ---------------------------------------------------------------------------
// 4) top-32 per row: per-thread sorted insertion + 32 block-argmax rounds.
//    key = (sortable(float) << 32) | (~col) so larger key = larger sim,
//    ties -> smaller index; keys are unique.
// ---------------------------------------------------------------------------
__global__ void topk_kernel(const float* __restrict__ sim, int* __restrict__ idx) {
    __shared__ unsigned long long cand[128 * 32];
    __shared__ unsigned long long redv[128];
    __shared__ int redp[128];
    const int b = blockIdx.x, t = threadIdx.x, base = t * 32;

    for (int j = 0; j < 32; ++j) cand[base + j] = 0ULL;
    const float* row = sim + (long long)b * NK;
    for (int c = t; c < NK; c += 128) {
        unsigned s = __float_as_uint(row[c]);
        s = (s & 0x80000000u) ? ~s : (s | 0x80000000u);
        unsigned long long p = ((unsigned long long)s << 32) |
                               (unsigned long long)(0xFFFFFFFFu - (unsigned)c);
        if (p > cand[base]) {               // beats current min -> insert sorted
            int pos = 0;
            while (pos < 31 && cand[base + pos + 1] < p) {
                cand[base + pos] = cand[base + pos + 1]; ++pos;
            }
            cand[base + pos] = p;
        }
    }
    __syncthreads();

    for (int r = 0; r < KK; ++r) {
        unsigned long long mv = 0ULL; int mp = base;
        for (int j = 0; j < 32; ++j) {
            unsigned long long v = cand[base + j];
            if (v > mv) { mv = v; mp = base + j; }
        }
        redv[t] = mv; redp[t] = mp; __syncthreads();
        for (int s = 64; s > 0; s >>= 1) {
            if (t < s && redv[t + s] > redv[t]) { redv[t] = redv[t + s]; redp[t] = redp[t + s]; }
            __syncthreads();
        }
        if (t == 0) {
            idx[b * KK + r] = (int)(0xFFFFFFFFu - (unsigned)(redv[0] & 0xFFFFFFFFu));
            cand[redp[0]] = 0ULL;
        }
        __syncthreads();
    }
}

// ---------------------------------------------------------------------------
// 5) gather selected keys -> f16 rows
// ---------------------------------------------------------------------------
__global__ void gather_f16_kernel(const float* __restrict__ mk,
                                  const int* __restrict__ idx,
                                  _Float16* __restrict__ mkh) {
    const int rk = blockIdx.x;            // 0..B*K-1
    const int r  = idx[rk];
    for (int j = threadIdx.x; j < DD; j += 128)
        mkh[(long long)rk * DD + j] = (_Float16)mk[(long long)r * DD + j];
}

__global__ void zero_att_kernel(float* __restrict__ att) {
    const int i = blockIdx.x * blockDim.x + threadIdx.x;
    #pragma unroll
    for (int j = 0; j < 4; ++j) att[i * 4 + j] = 0.f;
}

// ---------------------------------------------------------------------------
// 6) qt = relu(q @ Wq + bq)   (tiny: 33.5M MACs)
// ---------------------------------------------------------------------------
__global__ void qt_kernel(const float* __restrict__ qrelu,
                          const float* __restrict__ Wq,
                          const float* __restrict__ bq,
                          float* __restrict__ qt) {
    __shared__ float qrow[DD];
    const int b = blockIdx.x, t = threadIdx.x;
    qrow[t] = qrelu[b * DD + t];
    qrow[t + 256] = qrelu[b * DD + t + 256];
    __syncthreads();
    float acc = bq[t];
    for (int d = 0; d < DD; ++d) acc += qrow[d] * Wq[d * AU_ + t];
    qt[b * AU_ + t] = fmaxf(acc, 0.f);
}

// ---------------------------------------------------------------------------
// 7) mt = relu(mkh @ Wm + bm); attended[b] += mt  (softmax over size-1 dim == 1)
//    WMMA GEMM [8192 x 512] x [512 x 256], epilogue atomic-accumulates over k.
// ---------------------------------------------------------------------------
__global__ void mt_gemm_kernel(const _Float16* __restrict__ A,
                               const float* __restrict__ Wm,
                               const float* __restrict__ bm,
                               float* __restrict__ att) {
    __shared__ __attribute__((aligned(64))) _Float16 As[128 * 32]; // [m][k]
    __shared__ __attribute__((aligned(64))) _Float16 Bs[64 * 32];  // [n][k]
    const int t = threadIdx.x, lane = t & 31, wave = t >> 5;
    const int wr = wave & 3, wc = wave >> 2;           // 4x2 wave grid
    const int m0 = blockIdx.x * 128, n0 = blockIdx.y * 64;

    v8f acc[2][2] = {};

    for (int k0 = 0; k0 < DD; k0 += 32) {
        __syncthreads();
        {   // A tile: 128x32 halves, 16 halves per thread
            const int r = t >> 1, ko = (t & 1) * 16;
            const v8h* s = (const v8h*)(A + (long long)(m0 + r) * DD + k0 + ko);
            v8h* d = (v8h*)(As + r * 32 + ko);
            d[0] = s[0]; d[1] = s[1];
        }
        {   // B tile: transpose Wm[k][n] -> Bs[n][k], f32->f16
            const int kk = t >> 3, n8 = (t & 7) * 8;
            float4 v0 = *(const float4*)(Wm + (k0 + kk) * AU_ + n0 + n8);
            float4 v1 = *(const float4*)(Wm + (k0 + kk) * AU_ + n0 + n8 + 4);
            Bs[(n8 + 0) * 32 + kk] = (_Float16)v0.x;
            Bs[(n8 + 1) * 32 + kk] = (_Float16)v0.y;
            Bs[(n8 + 2) * 32 + kk] = (_Float16)v0.z;
            Bs[(n8 + 3) * 32 + kk] = (_Float16)v0.w;
            Bs[(n8 + 4) * 32 + kk] = (_Float16)v1.x;
            Bs[(n8 + 5) * 32 + kk] = (_Float16)v1.y;
            Bs[(n8 + 6) * 32 + kk] = (_Float16)v1.z;
            Bs[(n8 + 7) * 32 + kk] = (_Float16)v1.w;
        }
        __syncthreads();
        #pragma unroll
        for (int rt = 0; rt < 2; ++rt) {
            v16h a = load_a_frag(As, wr * 32 + rt * 16, 32, lane);
            #pragma unroll
            for (int ct = 0; ct < 2; ++ct) {
                v16h b = load_b_frag(Bs, wc * 32 + ct * 16, 32, lane);
                acc[rt][ct] = __builtin_amdgcn_wmma_f32_16x16x32_f16(
                    false, a, false, b, (short)0, acc[rt][ct], false, false);
            }
        }
    }

    const int nn = lane & 15, hi = lane >> 4;
    #pragma unroll
    for (int rt = 0; rt < 2; ++rt)
        #pragma unroll
        for (int ct = 0; ct < 2; ++ct)
            #pragma unroll
            for (int v = 0; v < 8; ++v) {
                const int m = m0 + wr * 32 + rt * 16 + hi * 8 + v;  // row in [0,8192)
                const int n = n0 + wc * 32 + ct * 16 + nn;
                float val = fmaxf(acc[rt][ct][v] + bm[n], 0.f);
                atomicAdd(att + (m >> 5) * AU_ + n, val);           // sum over k
            }
}

// ---------------------------------------------------------------------------
// 8) layernorm(attended + qt) then [q, ma] @ Wc + bc
// ---------------------------------------------------------------------------
__global__ void final_kernel(const float* __restrict__ att,
                             const float* __restrict__ qt,
                             const float* __restrict__ gamma,
                             const float* __restrict__ beta,
                             const float* __restrict__ qrelu,
                             const float* __restrict__ Wc,
                             const float* __restrict__ bc,
                             float* __restrict__ out) {
    __shared__ float ma[AU_], qrow[DD], red[256];
    const int b = blockIdx.x, t = threadIdx.x;
    const float x = att[b * AU_ + t] + qt[b * AU_ + t];
    red[t] = x; __syncthreads();
    for (int s = 128; s > 0; s >>= 1) { if (t < s) red[t] += red[t + s]; __syncthreads(); }
    const float mu = red[0] / (float)AU_; __syncthreads();
    const float dv = x - mu;
    red[t] = dv * dv; __syncthreads();
    for (int s = 128; s > 0; s >>= 1) { if (t < s) red[t] += red[t + s]; __syncthreads(); }
    const float var = red[0] / (float)AU_;
    ma[t] = dv * rsqrtf(var + 1e-5f) * gamma[t] + beta[t];
    qrow[t] = qrelu[b * DD + t];
    qrow[t + 256] = qrelu[b * DD + t + 256];
    __syncthreads();
    if (t < CC) {
        float acc = bc[t];
        for (int d = 0; d < DD; ++d)  acc += qrow[d] * Wc[d * CC + t];
        for (int a = 0; a < AU_; ++a) acc += ma[a] * Wc[(DD + a) * CC + t];
        out[b * CC + t] = acc;
    }
}

// ---------------------------------------------------------------------------
// Launch
// ---------------------------------------------------------------------------
extern "C" void kernel_launch(void* const* d_in, const int* in_sizes, int n_in,
                              void* d_out, int out_size, void* d_ws, size_t ws_size,
                              hipStream_t stream) {
    (void)in_sizes; (void)n_in; (void)out_size; (void)ws_size;
    const float* query = (const float*)d_in[0];
    const float* mk    = (const float*)d_in[1];
    const float* Wq    = (const float*)d_in[2];
    const float* bq    = (const float*)d_in[3];
    const float* Wm    = (const float*)d_in[4];
    const float* bm    = (const float*)d_in[5];
    const float* gamma = (const float*)d_in[6];
    const float* beta  = (const float*)d_in[7];
    const float* Wc    = (const float*)d_in[8];
    const float* bc    = (const float*)d_in[9];
    float* out = (float*)d_out;

    // workspace layout (bytes)
    char* ws = (char*)d_ws;
    float*    qrelu = (float*)   (ws);                       // 256*512*4   = 512 KB
    _Float16* qh    = (_Float16*)(ws + 524288);              // 256*512*2   = 256 KB
    float*    qn    = (float*)   (ws + 786432);              // 256*4
    float*    mn    = (float*)   (ws + 787456);              // 100000*4
    float*    sim   = (float*)   (ws + 1187840);             // 256*100000*4 = 102.4 MB
    int*      idx   = (int*)     (ws + 1187840 + 102400000); // 8192*4
    _Float16* mkh   = (_Float16*)(ws + 103620608);           // 8192*512*2  = 8 MB
    float*    qt    = (float*)   (ws + 112009216);           // 256*256*4
    float*    att   = (float*)   (ws + 112271360);           // 256*256*4

    prep_q_kernel   <<<BQ, 128, 0, stream>>>(query, qrelu, qh, qn);
    key_norms_kernel<<<512, 256, 0, stream>>>(mk, mn);
    sim_gemm_kernel <<<NK / 32, 256, 0, stream>>>(qh, mk, qn, mn, sim);
    topk_kernel     <<<BQ, 128, 0, stream>>>(sim, idx);
    gather_f16_kernel<<<BQ * KK, 128, 0, stream>>>(mk, idx, mkh);
    zero_att_kernel <<<64, 256, 0, stream>>>(att);
    qt_kernel       <<<BQ, 256, 0, stream>>>(qrelu, Wq, bq, qt);
    mt_gemm_kernel  <<<dim3((BQ * KK) / 128, AU_ / 64), 256, 0, stream>>>(mkh, Wm, bm, att);
    final_kernel    <<<BQ, 256, 0, stream>>>(att, qt, gamma, beta, qrelu, Wc, bc, out);
}
